// BahdanauAttention_34265249087977
// MI455X (gfx1250) — matrix operand order repaired and checked by reading, not compiled, not run on hardware
//
#include <hip/hip_runtime.h>
#include <hip/hip_bf16.h>
#include <math.h>

#define B_ 32
#define T_ 2048
#define H_ 1024
#define F_ 1024

typedef __attribute__((ext_vector_type(16))) __bf16 v16bf;
typedef __attribute__((ext_vector_type(8)))  float  v8f;

union FragBF { v16bf f; uint4 u[2]; };

__device__ __forceinline__ uint2 pack_bf16x4(float4 v) {
    union { __bf16 h[4]; uint2 u; } p;
    p.h[0] = (__bf16)v.x; p.h[1] = (__bf16)v.y;
    p.h[2] = (__bf16)v.z; p.h[3] = (__bf16)v.w;
    return p.u;
}

// ---------------------------------------------------------------------------
// 1) qb[b,o] = sum_h query[b,h]*Wa_w[o,h] + Wa_b[o] + Ua_b[o]
// ---------------------------------------------------------------------------
__global__ __launch_bounds__(256) void qb_kernel(
    const float* __restrict__ query, const float* __restrict__ Wa_w,
    const float* __restrict__ Wa_b, const float* __restrict__ Ua_b,
    float* __restrict__ qb)
{
    const int o = blockIdx.x * 256 + threadIdx.x;
    const int b = blockIdx.y;
    const float4* qp = reinterpret_cast<const float4*>(query + (size_t)b * H_);
    const float4* wp = reinterpret_cast<const float4*>(Wa_w + (size_t)o * H_);
    float acc = 0.f;
#pragma unroll 4
    for (int i = 0; i < H_ / 4; ++i) {
        float4 q4 = qp[i], w4 = wp[i];
        acc += q4.x * w4.x + q4.y * w4.y + q4.z * w4.z + q4.w * w4.w;
    }
    qb[(size_t)b * H_ + o] = acc + Wa_b[o] + Ua_b[o];
}

// ---------------------------------------------------------------------------
// 2) Ua_w f32 -> bf16 (row-major, H x F)
// ---------------------------------------------------------------------------
__global__ __launch_bounds__(256) void cvt_kernel(
    const float* __restrict__ Ua_w, __bf16* __restrict__ ua_bf)
{
    const int i = blockIdx.x * 256 + threadIdx.x;      // float4 index
    float4 v = reinterpret_cast<const float4*>(Ua_w)[i];
    reinterpret_cast<uint2*>(ua_bf)[i] = pack_bf16x4(v);
}

// ---------------------------------------------------------------------------
// 3) Fused: k = keys@Ua^T ; e = tanh(qb + k) ; scores = e . Va + Va_b
//    One WG per (b, 64 t-rows). 16 waves; wave tile M=64 x N=64
//    (4x4 accumulator tiles). B frags double-buffered against WMMA.
// ---------------------------------------------------------------------------
#define LDS_STRIDE 1032           // 1024 + 8 bf16 pad: kills LDS bank conflicts
#define MT_STRIDE  (16 * LDS_STRIDE)   // 16 rows apart, immediate-addressable

__global__ __launch_bounds__(512) void score_kernel(
    const float*  __restrict__ keys, const __bf16* __restrict__ ua_bf,
    const float*  __restrict__ qb,   const float*  __restrict__ va,
    const float*  __restrict__ va_b, float* __restrict__ scores)
{
    __shared__ unsigned short ldsk[64 * LDS_STRIDE];   // ~132 KB bf16 keys tile
    __shared__ float lds_sc[64];

    const int tid = threadIdx.x;
    const int b   = blockIdx.y;
    const int t0  = blockIdx.x * 64;

    // Stage keys tile (64 x 1024) into LDS as bf16
    {
        const float* kb = keys + ((size_t)b * T_ + t0) * F_;
        for (int i = tid; i < 64 * 256; i += 512) {            // 16384 float4
            const int row = i >> 8;
            const int c4  = i & 255;
            float4 v = *reinterpret_cast<const float4*>(kb + (size_t)row * F_ + c4 * 4);
            *reinterpret_cast<uint2*>(&ldsk[row * LDS_STRIDE + c4 * 4]) = pack_bf16x4(v);
        }
    }
    if (tid < 64) lds_sc[tid] = 0.f;
    __syncthreads();

    const int wave  = tid >> 5;
    const int lane  = tid & 31;
    const int half  = lane >> 4;    // K-half select (this code's contiguous k = 16*half + s order)
    const int l15   = lane & 15;
    const int nbase = wave * 64;    // 16 waves x 64 = 1024 o-columns

    v8f acc[4][4];
    const v8f vzero = {0.f, 0.f, 0.f, 0.f, 0.f, 0.f, 0.f, 0.f};
#pragma unroll
    for (int mt = 0; mt < 4; ++mt)
#pragma unroll
        for (int nt = 0; nt < 4; ++nt) acc[mt][nt] = vzero;

    // A: lane l (both halves) holds row M=l; K contiguous 0-15 / 16-31 per half.
    // Two base pointers (rows 0-31 / rows 32-63). The empty asm makes off_hi
    // opaque so the compiler keeps a dedicated loop-carried address register
    // instead of re-deriving (base + 0x10200) with per-iteration VALU adds
    // that alias WMMA source registers (which costs 4 co-exec v_nops each).
    unsigned off_lo = (unsigned)((     l15) * LDS_STRIDE + half * 16);
    unsigned off_hi = (unsigned)((32 + l15) * LDS_STRIDE + half * 16);
    asm volatile("" : "+v"(off_hi));
    const unsigned short* a_lo = &ldsk[off_lo];
    const unsigned short* a_hi = &ldsk[off_hi];
    // B: lane l holds column N=l = Ua row; K contiguous 0-15 / 16-31 per half
    const __bf16* bbase = ua_bf + (size_t)(nbase + l15) * F_ + half * 16;

    FragBF bcur[4], bnext[4];
#pragma unroll
    for (int nt = 0; nt < 4; ++nt) {
        const __bf16* p = bbase + (size_t)nt * 16 * F_;
        bcur[nt].u[0] = *reinterpret_cast<const uint4*>(p);
        bcur[nt].u[1] = *reinterpret_cast<const uint4*>(p + 8);
    }

    for (int k0 = 0; k0 < F_; k0 += 64) {
        // prefetch B for k0+32 while computing with bcur (k0)
#pragma unroll
        for (int nt = 0; nt < 4; ++nt) {
            const __bf16* p = bbase + (size_t)nt * 16 * F_ + (k0 + 32);
            bnext[nt].u[0] = *reinterpret_cast<const uint4*>(p);
            bnext[nt].u[1] = *reinterpret_cast<const uint4*>(p + 8);
        }
#pragma unroll
        for (int mt = 0; mt < 4; ++mt) {
            const unsigned short* ab = (mt < 2) ? a_lo : a_hi;
            const int ro = (mt & 1) * MT_STRIDE;
            FragBF a;
            a.u[0] = *reinterpret_cast<const uint4*>(ab + ro + k0);
            a.u[1] = *reinterpret_cast<const uint4*>(ab + ro + k0 + 8);
#pragma unroll
            for (int nt = 0; nt < 4; ++nt)
                acc[mt][nt] = __builtin_amdgcn_wmma_f32_16x16x32_bf16(
                    false, a.f, false, bcur[nt].f, (short)0, acc[mt][nt], false, false);
        }
        // prefetch B for k0+64 (tail read is unused garbage inside our ws)
#pragma unroll
        for (int nt = 0; nt < 4; ++nt) {
            const __bf16* p = bbase + (size_t)nt * 16 * F_ + (k0 + 64);
            bcur[nt].u[0] = *reinterpret_cast<const uint4*>(p);
            bcur[nt].u[1] = *reinterpret_cast<const uint4*>(p + 8);
        }
#pragma unroll
        for (int mt = 0; mt < 4; ++mt) {
            const unsigned short* ab = (mt < 2) ? a_lo : a_hi;
            const int ro = (mt & 1) * MT_STRIDE;
            FragBF a;
            a.u[0] = *reinterpret_cast<const uint4*>(ab + ro + k0 + 32);
            a.u[1] = *reinterpret_cast<const uint4*>(ab + ro + k0 + 40);
#pragma unroll
            for (int nt = 0; nt < 4; ++nt)
                acc[mt][nt] = __builtin_amdgcn_wmma_f32_16x16x32_bf16(
                    false, a.f, false, bnext[nt].f, (short)0, acc[mt][nt], false, false);
        }
    }

    // Epilogue: tanh(C + qb) . Va, reduce over this wave's 64 o-columns
    float qv[4], vv[4];
#pragma unroll
    for (int nt = 0; nt < 4; ++nt) {
        const int o = nbase + nt * 16 + l15;
        qv[nt] = qb[(size_t)b * H_ + o];
        vv[nt] = va[o];
    }
#pragma unroll
    for (int mt = 0; mt < 4; ++mt) {
#pragma unroll
        for (int j = 0; j < 8; ++j) {
            // C/D layout: VGPR j, lanes 0-15 -> M=j ; lanes 16-31 -> M=j+8
            float s = 0.f;
#pragma unroll
            for (int nt = 0; nt < 4; ++nt)
                s += vv[nt] * tanhf(acc[mt][nt][j] + qv[nt]);
            // reduce across the 16 lanes of each half-wave (N columns)
            s += __shfl_xor(s, 1);
            s += __shfl_xor(s, 2);
            s += __shfl_xor(s, 4);
            s += __shfl_xor(s, 8);
            if (l15 == 0)
                atomicAdd(&lds_sc[mt * 16 + half * 8 + j], s);
        }
    }
    __syncthreads();
    if (tid < 64)
        scores[(size_t)b * T_ + t0 + tid] = lds_sc[tid] + va_b[0];
}

// ---------------------------------------------------------------------------
// 4) softmax over T per batch; weights straight into d_out
// ---------------------------------------------------------------------------
__global__ __launch_bounds__(256) void softmax_kernel(
    const float* __restrict__ scores, float* __restrict__ out_w)
{
    __shared__ float red[256];
    const int b = blockIdx.x, tid = threadIdx.x;
    const float* sb = scores + (size_t)b * T_;
    float sv[8];
    float m = -3.4e38f;
#pragma unroll
    for (int j = 0; j < 8; ++j) { sv[j] = sb[tid + j * 256]; m = fmaxf(m, sv[j]); }
    red[tid] = m; __syncthreads();
    for (int s = 128; s > 0; s >>= 1) {
        if (tid < s) red[tid] = fmaxf(red[tid], red[tid + s]);
        __syncthreads();
    }
    m = red[0]; __syncthreads();
    float sum = 0.f;
#pragma unroll
    for (int j = 0; j < 8; ++j) { sv[j] = __expf(sv[j] - m); sum += sv[j]; }
    red[tid] = sum; __syncthreads();
    for (int s = 128; s > 0; s >>= 1) {
        if (tid < s) red[tid] += red[tid + s];
        __syncthreads();
    }
    const float inv = 1.f / red[0];
    float* ob = out_w + (size_t)b * T_;
#pragma unroll
    for (int j = 0; j < 8; ++j) ob[tid + j * 256] = sv[j] * inv;
}

// ---------------------------------------------------------------------------
// 5) context[b,f] = sum_t w[b,t] * keys[b,t,f]  (HBM-streaming, coalesced)
// ---------------------------------------------------------------------------
__global__ __launch_bounds__(256) void context_kernel(
    const float* __restrict__ keys, const float* __restrict__ w,
    float* __restrict__ ctx)
{
    __shared__ float  wsh[T_];
    __shared__ float4 red4[256];
    const int tid   = threadIdx.x;
    const int b     = blockIdx.y;
    const int fbase = blockIdx.x * 256;
#pragma unroll
    for (int j = 0; j < 8; ++j)
        wsh[tid + j * 256] = w[(size_t)b * T_ + tid + j * 256];
    __syncthreads();

    const int fi = (tid & 63) << 2;   // float4 column within 256-chunk
    const int tg = tid >> 6;          // 4-way t partition
    float4 acc = make_float4(0.f, 0.f, 0.f, 0.f);
    const float* kb = keys + (size_t)b * T_ * F_ + fbase + fi;
#pragma unroll 4
    for (int t = tg; t < T_; t += 4) {
        if (t + 16 < T_)
            __builtin_prefetch(kb + (size_t)(t + 16) * F_, 0, 0);  // global_prefetch_b8
        float4 kv = *reinterpret_cast<const float4*>(kb + (size_t)t * F_);
        const float ww = wsh[t];
        acc.x += ww * kv.x; acc.y += ww * kv.y;
        acc.z += ww * kv.z; acc.w += ww * kv.w;
    }
    red4[tid] = acc; __syncthreads();
    if (tid < 64) {
        float4 r  = red4[tid];
        float4 r1 = red4[tid + 64], r2 = red4[tid + 128], r3 = red4[tid + 192];
        r.x += r1.x + r2.x + r3.x; r.y += r1.y + r2.y + r3.y;
        r.z += r1.z + r2.z + r3.z; r.w += r1.w + r2.w + r3.w;
        *reinterpret_cast<float4*>(ctx + (size_t)b * F_ + fbase + tid * 4) = r;
    }
}

// ---------------------------------------------------------------------------
extern "C" void kernel_launch(void* const* d_in, const int* in_sizes, int n_in,
                              void* d_out, int out_size, void* d_ws, size_t ws_size,
                              hipStream_t stream)
{
    const float* query = (const float*)d_in[0];   // (B,1,H)
    const float* keys  = (const float*)d_in[1];   // (B,T,F)
    const float* Wa_w  = (const float*)d_in[2];   // (H,H)
    const float* Wa_b  = (const float*)d_in[3];   // (H,)
    const float* Ua_w  = (const float*)d_in[4];   // (H,F)
    const float* Ua_b  = (const float*)d_in[5];   // (H,)
    const float* Va_w  = (const float*)d_in[6];   // (1,H)
    const float* Va_b  = (const float*)d_in[7];   // (1,)

    float* out_ctx = (float*)d_out;                        // (B,1,F)
    float* out_w   = out_ctx + (size_t)B_ * F_;            // (B,1,T)

    char*   ws     = (char*)d_ws;
    float*  qb     = (float*)ws;                           // B*H f32   = 128 KB
    __bf16* ua_bf  = (__bf16*)(ws + (size_t)B_ * H_ * 4);  // H*F bf16  =   2 MB
    float*  scores = (float*)(ws + (size_t)B_ * H_ * 4
                                 + (size_t)H_ * F_ * 2);   // B*T f32   = 256 KB

    qb_kernel     <<<dim3(H_ / 256, B_), 256, 0, stream>>>(query, Wa_w, Wa_b, Ua_b, qb);
    cvt_kernel    <<<dim3((H_ * F_) / 1024), 256, 0, stream>>>(Ua_w, ua_bf);
    score_kernel  <<<dim3(T_ / 64, B_), 512, 0, stream>>>(keys, ua_bf, qb, Va_w, Va_b, scores);
    softmax_kernel<<<dim3(B_), 256, 0, stream>>>(scores, out_w);
    context_kernel<<<dim3(F_ / 256, B_), 256, 0, stream>>>(keys, out_w, out_ctx);
}
